// MLMModel_42262478193314
// MI455X (gfx1250) — matrix operand order repaired and checked by reading, not compiled, not run on hardware
//
#include <hip/hip_runtime.h>
#include <math.h>

// ---------------- model dimensions ----------------
constexpr int Bb = 4, Tt = 512, Dd = 768, Hh = 12, Ll = 4, Vv = 30522;
constexpr int DKk = Dd / Hh;       // 64
constexpr int FFf = 4 * Dd;        // 3072
constexpr int ROWS = Bb * Tt;      // 2048

// ---------------- vector / WMMA types (CDNA5, wave32) ----------------
typedef __attribute__((ext_vector_type(16))) __bf16        v16bf;
typedef __attribute__((ext_vector_type(2)))  __bf16        v2bf;
typedef __attribute__((ext_vector_type(8)))  float         v8f;
typedef __attribute__((ext_vector_type(4)))  float         v4f;
typedef __attribute__((ext_vector_type(2)))  float         v2f;
typedef __attribute__((ext_vector_type(4)))  unsigned int  v4u;

union FragU { v16bf v; v4u q[2]; };

// Hardware bf16 conversion (v_cvt_*bf16_f32) instead of manual RNE bit math.
__device__ __forceinline__ unsigned int pk2(float lo, float hi) {
  union { v2bf b; unsigned int u; } c;
  c.b[0] = (__bf16)lo;
  c.b[1] = (__bf16)hi;
  return c.u;
}

// Tile geometry: 256 threads = 8 waves stacked in M; block tile 256x64;
// wave tile 32x64 (2x4 of 16x16) -> 8 WMMAs share 4 B-fragments per K-step.
#define BM 256
#define BN 64
#define BK 32
// LDS row stride = 20 dwords (40 bf16, 80 bytes): 16B-aligned rows, and the
// stride-20 lane->bank map gives conflict-free ds_load_b128 fragment reads.
#define LSTR 20

// A tile in LDS: [m][k], m-major, k packed 2/dword.
// A fragment (16x32): lanes 0-15 row m=lane, k 0..7 & 16..23;
//                     lanes 16-31 row m=lane-16, k 8..15 & 24..31.
__device__ __forceinline__ v16bf ldfragA(const unsigned int* Asu, int row0, int lane) {
  int m = row0 + (lane & 15);
  int s = (lane < 16) ? 0 : 4;           // dword offset: k 0 / k 8
  FragU f;
  f.q[0] = *(const v4u*)(Asu + m * LSTR + s);
  f.q[1] = *(const v4u*)(Asu + m * LSTR + 8 + s);
  return f.v;
}

// B tile in LDS: [n][k] (n-major, i.e. transposed), k packed 2/dword.
// B fragment (32x16): lane n<16 column n, k 0..7 & 16..23; lane>=16 column n-16, k 8..15 & 24..31.
__device__ __forceinline__ v16bf ldfragB(const unsigned int* Bsu, int col0, int lane) {
  int n = col0 + (lane & 15);
  int s = (lane < 16) ? 0 : 4;           // dword offset: k 0 / k 8
  FragU f;
  f.q[0] = *(const v4u*)(Bsu + n * LSTR + s);
  f.q[1] = *(const v4u*)(Bsu + n * LSTR + s + 8);
  return f.v;
}

#define WMMA_BF16(ACC, Afr, Bfr) \
  __builtin_amdgcn_wmma_f32_16x16x32_bf16(false, (Afr), false, (Bfr), (short)0, (ACC), false, false)

#define WMMA_OCT()                                                    \
  do {                                                                \
    v16bf a0 = ldfragA(Asu, wm,      lane);                           \
    v16bf a1 = ldfragA(Asu, wm + 16, lane);                           \
    v16bf b0 = ldfragB(Bsu, 0,  lane);                                \
    v16bf b1 = ldfragB(Bsu, 16, lane);                                \
    v16bf b2 = ldfragB(Bsu, 32, lane);                                \
    v16bf b3 = ldfragB(Bsu, 48, lane);                                \
    acc[0][0] = WMMA_BF16(acc[0][0], a0, b0);                         \
    acc[0][1] = WMMA_BF16(acc[0][1], a0, b1);                         \
    acc[0][2] = WMMA_BF16(acc[0][2], a0, b2);                         \
    acc[0][3] = WMMA_BF16(acc[0][3], a0, b3);                         \
    acc[1][0] = WMMA_BF16(acc[1][0], a1, b0);                         \
    acc[1][1] = WMMA_BF16(acc[1][1], a1, b1);                         \
    acc[1][2] = WMMA_BF16(acc[1][2], a1, b2);                         \
    acc[1][3] = WMMA_BF16(acc[1][3], a1, b3);                         \
  } while (0)

// Stage A tile (256 rows x 32 k): 4 threads/row x 8 floats, 4 passes.
// Coalesced 2x float4 global loads + one 16B-aligned ds_store_b128 per pass.
// ap = &A[m0 + rr][k0 + kc], rowStep = 64*lda, aw = Asu + rr*LSTR + (kc>>1).
__device__ __forceinline__ void stageA(const float* ap, size_t rowStep,
                                       unsigned int* aw) {
#pragma unroll
  for (int p = 0; p < 4; ++p) {
    v4f x0 = *(const v4f*)(ap);
    v4f x1 = *(const v4f*)(ap + 4);
    v4u pk;
    pk[0] = pk2(x0[0], x0[1]);
    pk[1] = pk2(x0[2], x0[3]);
    pk[2] = pk2(x1[0], x1[1]);
    pk[3] = pk2(x1[2], x1[3]);
    *(v4u*)(aw) = pk;
    ap += rowStep;
    aw += 64 * LSTR;
  }
}

// ---------------------------------------------------------------------------
// Generic GEMM: out[M,N] = A[M,K] @ B[K,N]  (fp32 global -> bf16 LDS -> WMMA)
// Requires: M % 256 == 0, K % 32 == 0 (true for every call site). N may be ragged.
// mode 0: out = acc (+bias)    mode 1: out = gelu(acc+bias)
// mode 2: out = acc + bias + resid   (in-place residual supported)
// ---------------------------------------------------------------------------
__global__ __launch_bounds__(256) void gemm_wmma(
    const float* __restrict__ A, int lda,
    const float* __restrict__ Bm, int ldb,
    const float* __restrict__ bias,
    const float* __restrict__ resid,
    float* __restrict__ out, int ldo,
    int N, int K, int mode)
{
  __shared__ __align__(16) unsigned int Asu[BM * LSTR];
  __shared__ __align__(16) unsigned int Bsu[BN * LSTR];
  const int tid  = threadIdx.x;
  const int lane = tid & 31;
  const int wm   = (tid >> 5) * 32;        // wave M offset 0..224
  const int m0 = blockIdx.y * BM;
  const int n0 = blockIdx.x * BN;

  v8f acc[2][4] = {};

  // A stage: 4 threads per row, 8 floats each
  const int rr = tid >> 2;                 // row 0..63 (per pass)
  const int kc = (tid & 3) * 8;            // k chunk
  // B stage: thread -> (n chunk of 4, k pair kp): reads rows 2kp,2kp+1
  const int nc = tid & 15;
  const int kp = tid >> 4;

  const float* aptr  = A + (size_t)(m0 + rr) * lda + kc;
  const size_t arow  = (size_t)64 * lda;
  const float* bptr0 = Bm + (size_t)(2 * kp) * ldb + (n0 + nc * 4);
  const float* bptr1 = bptr0 + ldb;
  unsigned int* asw = Asu + rr * LSTR + (kc >> 1);
  unsigned int* bsw = Bsu + (nc * 4) * LSTR + kp;
  const int nrem = N - (n0 + nc * 4);
  const size_t bstep = (size_t)BK * ldb;

  for (int k0 = 0; k0 < K; k0 += BK) {
    stageA(aptr, arow, asw);
    if (nrem >= 4) {
      v2f r0a = *(const v2f*)(bptr0);
      v2f r0b = *(const v2f*)(bptr0 + 2);
      v2f r1a = *(const v2f*)(bptr1);
      v2f r1b = *(const v2f*)(bptr1 + 2);
      bsw[0 * LSTR] = pk2(r0a[0], r1a[0]);
      bsw[1 * LSTR] = pk2(r0a[1], r1a[1]);
      bsw[2 * LSTR] = pk2(r0b[0], r1b[0]);
      bsw[3 * LSTR] = pk2(r0b[1], r1b[1]);
    } else {
      float t0v[4] = {0.f, 0.f, 0.f, 0.f};
      float t1v[4] = {0.f, 0.f, 0.f, 0.f};
      if (nrem > 0) {
        for (int i = 0; i < nrem && i < 4; ++i) { t0v[i] = bptr0[i]; t1v[i] = bptr1[i]; }
      }
#pragma unroll
      for (int j = 0; j < 4; ++j) bsw[j * LSTR] = pk2(t0v[j], t1v[j]);
    }
    if (k0 + BK < K && nrem > 0)
      __builtin_prefetch(bptr0 + bstep, 0, 1);   // global_prefetch next B tile
    __syncthreads();
    WMMA_OCT();
    __syncthreads();
    aptr += BK;
    bptr0 += bstep;
    bptr1 += bstep;
  }

  const int en = lane & 15;
  const int em = (lane < 16) ? 0 : 8;
#pragma unroll
  for (int ti = 0; ti < 2; ++ti) {
#pragma unroll
    for (int tj = 0; tj < 4; ++tj) {
      int gmb = m0 + wm + ti * 16 + em;
      int gn  = n0 + tj * 16 + en;
      if (gn >= N) continue;
      float bv = bias ? bias[gn] : 0.0f;
#pragma unroll
      for (int j = 0; j < 8; ++j) {
        int gm = gmb + j;
        float v = acc[ti][tj][j] + bv;
        if (mode == 1) {
          v = 0.5f * v * (1.0f + erff(v * 0.70710678118654752440f));
        } else if (mode == 2) {
          v += resid[(size_t)gm * ldo + gn];
        }
        out[(size_t)gm * ldo + gn] = v;
      }
    }
  }
}

// ---------------------------------------------------------------------------
// Angular scores: per (b,h): S[t,s] = f(Qn.Kn), f(c)=(max(1-acos(clip c)/pi,1e-6))^16
// grid = (T/BN, T/BM, B*H). LDS B tile is [s][d] -> contiguous reads AND writes.
// ---------------------------------------------------------------------------
__global__ __launch_bounds__(256) void attn_scores(
    const float* __restrict__ q, const float* __restrict__ k,
    float* __restrict__ scores)
{
  __shared__ __align__(16) unsigned int Asu[BM * LSTR];
  __shared__ __align__(16) unsigned int Bsu[BN * LSTR];
  const int tid  = threadIdx.x;
  const int lane = tid & 31;
  const int wm   = (tid >> 5) * 32;
  const int z = blockIdx.z;
  const int b = z / Hh, h = z % Hh;
  const int t0 = blockIdx.y * BM;
  const int s0 = blockIdx.x * BN;

  v8f acc[2][4] = {};

  const int rr = tid >> 2;
  const int kc = (tid & 3) * 8;
  const int bs = tid >> 2;             // s row 0..63
  const int bd = (tid & 3) * 8;        // d chunk

  const float* aptr = q + ((size_t)(b * Tt + t0 + rr)) * Dd + h * DKk + kc;
  const size_t arow = (size_t)64 * Dd;
  const float* kptr = k + ((size_t)(b * Tt + s0 + bs)) * Dd + h * DKk + bd;
  unsigned int* asw = Asu + rr * LSTR + (kc >> 1);
  unsigned int* bsw = Bsu + bs * LSTR + (bd >> 1);

  for (int k0 = 0; k0 < DKk; k0 += BK) {   // 2 iterations
    stageA(aptr, arow, asw);
    {
      v4f y0 = *(const v4f*)(kptr);
      v4f y1 = *(const v4f*)(kptr + 4);
      v4u pk;
      pk[0] = pk2(y0[0], y0[1]);
      pk[1] = pk2(y0[2], y0[3]);
      pk[2] = pk2(y1[0], y1[1]);
      pk[3] = pk2(y1[2], y1[3]);
      *(v4u*)bsw = pk;
    }
    __syncthreads();
    WMMA_OCT();
    __syncthreads();
    aptr += BK;
    kptr += BK;
  }

  const float INV_PI = 0.31830988618379067154f;
  const int en = lane & 15;
  const int em = (lane < 16) ? 0 : 8;
  float* outz = scores + (size_t)z * Tt * Tt;
#pragma unroll
  for (int ti = 0; ti < 2; ++ti) {
#pragma unroll
    for (int tj = 0; tj < 4; ++tj) {
      int gmb = t0 + wm + ti * 16 + em;
      int gn  = s0 + tj * 16 + en;
#pragma unroll
      for (int j = 0; j < 8; ++j) {
        float c = acc[ti][tj][j];
        c = fminf(fmaxf(c, -0.999f), 0.999f);
        float sc = 1.0f - acosf(c) * INV_PI;
        float w = fmaxf(sc, 1e-6f);
        w = w * w; w = w * w; w = w * w; w = w * w;   // ^16
        outz[(size_t)(gmb + j) * Tt + gn] = w;
      }
    }
  }
}

// ---------------------------------------------------------------------------
// ctx[b,t,h*64+d] = sum_s Wm[z,t,s] * V[b,s,h*64+d]     grid = (1, T/BM, B*H)
// ---------------------------------------------------------------------------
__global__ __launch_bounds__(256) void attn_ctx(
    const float* __restrict__ wmn, const float* __restrict__ v,
    float* __restrict__ ctx)
{
  __shared__ __align__(16) unsigned int Asu[BM * LSTR];
  __shared__ __align__(16) unsigned int Bsu[BN * LSTR];
  const int tid  = threadIdx.x;
  const int lane = tid & 31;
  const int wm   = (tid >> 5) * 32;
  const int z = blockIdx.z;
  const int b = z / Hh, h = z % Hh;
  const int t0 = blockIdx.y * BM;

  v8f acc[2][4] = {};

  const int rr = tid >> 2;
  const int kc = (tid & 3) * 8;
  const int nc = tid & 15;
  const int kp = tid >> 4;

  const float* aptr  = wmn + (size_t)z * Tt * Tt + (size_t)(t0 + rr) * Tt + kc;
  const size_t arow  = (size_t)64 * Tt;
  const float* bptr0 = v + ((size_t)(b * Tt + 2 * kp)) * Dd + h * DKk + nc * 4;
  const float* bptr1 = bptr0 + Dd;
  unsigned int* asw = Asu + rr * LSTR + (kc >> 1);
  unsigned int* bsw = Bsu + (nc * 4) * LSTR + kp;
  const size_t bstep = (size_t)BK * Dd;

  for (int k0 = 0; k0 < Tt; k0 += BK) {   // 16 iterations over s
    stageA(aptr, arow, asw);
    {
      v4f r0 = *(const v4f*)(bptr0);
      v4f r1 = *(const v4f*)(bptr1);
      bsw[0 * LSTR] = pk2(r0[0], r1[0]);
      bsw[1 * LSTR] = pk2(r0[1], r1[1]);
      bsw[2 * LSTR] = pk2(r0[2], r1[2]);
      bsw[3 * LSTR] = pk2(r0[3], r1[3]);
    }
    __syncthreads();
    WMMA_OCT();
    __syncthreads();
    aptr += BK;
    bptr0 += bstep;
    bptr1 += bstep;
  }

  const int en = lane & 15;
  const int em = (lane < 16) ? 0 : 8;
#pragma unroll
  for (int ti = 0; ti < 2; ++ti) {
#pragma unroll
    for (int tj = 0; tj < 4; ++tj) {
      int gmb = t0 + wm + ti * 16 + em;
      int gn  = tj * 16 + en;              // 0..63 within head
#pragma unroll
      for (int j = 0; j < 8; ++j) {
        ctx[((size_t)(b * Tt + gmb + j)) * Dd + h * DKk + gn] = acc[ti][tj][j];
      }
    }
  }
}

// ---------------- elementwise / reduction kernels ----------------

__global__ void embed_kernel(const int* __restrict__ ids, const float* __restrict__ tok,
                             const float* __restrict__ pos, float* __restrict__ x, int total)
{
  int i = blockIdx.x * 256 + threadIdx.x;
  if (i >= total) return;
  int row = i / Dd, c = i - row * Dd;
  int t = row & (Tt - 1);
  x[i] = tok[(size_t)ids[row] * Dd + c] + pos[(size_t)t * Dd + c];
}

__global__ __launch_bounds__(256) void layernorm_kernel(
    const float* __restrict__ x, const float* __restrict__ g,
    const float* __restrict__ bb, float* __restrict__ y)
{
  __shared__ float rs[256], rq[256];
  int row = blockIdx.x, tid = threadIdx.x;
  const float* xr = x + (size_t)row * Dd;
  float v0 = xr[tid], v1 = xr[tid + 256], v2 = xr[tid + 512];
  rs[tid] = v0 + v1 + v2;
  rq[tid] = v0 * v0 + v1 * v1 + v2 * v2;
  __syncthreads();
  for (int s = 128; s > 0; s >>= 1) {
    if (tid < s) { rs[tid] += rs[tid + s]; rq[tid] += rq[tid + s]; }
    __syncthreads();
  }
  float mu  = rs[0] * (1.0f / Dd);
  float var = rq[0] * (1.0f / Dd) - mu * mu;
  float rstd = rsqrtf(var + 1e-5f);
  float* yr = y + (size_t)row * Dd;
  yr[tid]       = (v0 - mu) * rstd * g[tid]       + bb[tid];
  yr[tid + 256] = (v1 - mu) * rstd * g[tid + 256] + bb[tid + 256];
  yr[tid + 512] = (v2 - mu) * rstd * g[tid + 512] + bb[tid + 512];
}

__global__ void l2norm_qk(float* __restrict__ q, float* __restrict__ k, int nrh)
{
  int i = blockIdx.x * 256 + threadIdx.x;
  if (i >= nrh) return;
  int row = i / Hh, h = i - (i / Hh) * Hh;
  float* pq = q + (size_t)row * Dd + h * DKk;
  float* pk = k + (size_t)row * Dd + h * DKk;
  float sq = 0.0f, sk = 0.0f;
#pragma unroll 8
  for (int d = 0; d < DKk; ++d) { sq += pq[d] * pq[d]; sk += pk[d] * pk[d]; }
  float rq = 1.0f / fmaxf(sqrtf(sq), 1e-6f);
  float rk = 1.0f / fmaxf(sqrtf(sk), 1e-6f);
#pragma unroll 8
  for (int d = 0; d < DKk; ++d) { pq[d] *= rq; pk[d] *= rk; }
}

__global__ __launch_bounds__(256) void attn_norm_kernel(float* __restrict__ sc)
{
  __shared__ float red[256];
  float* row = sc + (size_t)blockIdx.x * Tt;
  int tid = threadIdx.x;
  float a = row[tid], b = row[tid + 256];
  red[tid] = a + b;
  __syncthreads();
  for (int s = 128; s > 0; s >>= 1) {
    if (tid < s) red[tid] += red[tid + s];
    __syncthreads();
  }
  float inv = 1.0f / (red[0] + 1e-6f);
  row[tid] = a * inv;
  row[tid + 256] = b * inv;
}

// ---------------------------------------------------------------------------
extern "C" void kernel_launch(void* const* d_in, const int* in_sizes, int n_in,
                              void* d_out, int out_size, void* d_ws, size_t ws_size,
                              hipStream_t stream)
{
  (void)in_sizes; (void)n_in; (void)out_size; (void)ws_size;
  const int*   ids   = (const int*)  d_in[0];
  const float* tok   = (const float*)d_in[1];
  const float* pos   = (const float*)d_in[2];
  const float* Wq    = (const float*)d_in[3];
  const float* Wk    = (const float*)d_in[4];
  const float* Wv    = (const float*)d_in[5];
  const float* Wo    = (const float*)d_in[6];
  const float* bo    = (const float*)d_in[7];
  const float* ln1g  = (const float*)d_in[8];
  const float* ln1b  = (const float*)d_in[9];
  const float* ln2g  = (const float*)d_in[10];
  const float* ln2b  = (const float*)d_in[11];
  const float* W1    = (const float*)d_in[12];
  const float* b1    = (const float*)d_in[13];
  const float* W2    = (const float*)d_in[14];
  const float* b2    = (const float*)d_in[15];
  const float* lnfg  = (const float*)d_in[16];
  const float* lnfb  = (const float*)d_in[17];
  const float* headw = (const float*)d_in[18];
  const float* headb = (const float*)d_in[19];
  float* out = (float*)d_out;

  char* ws = (char*)d_ws;
  size_t off = 0;
  auto take = [&](size_t bytes) -> char* {
    char* p = ws + off;
    off += (bytes + 255) & ~(size_t)255;
    return p;
  };
  float* xf  = (float*)take((size_t)ROWS * Dd * 4);   // residual stream
  float* xln = (float*)take((size_t)ROWS * Dd * 4);   // LN output
  float* qb  = (float*)take((size_t)ROWS * Dd * 4);
  float* kb  = (float*)take((size_t)ROWS * Dd * 4);
  float* vb  = (float*)take((size_t)ROWS * Dd * 4);
  float* cx  = (float*)take((size_t)ROWS * Dd * 4);
  float* mid = (float*)take((size_t)ROWS * FFf * 4);
  float* sc  = (float*)take((size_t)Bb * Hh * Tt * Tt * 4);

  embed_kernel<<<(ROWS * Dd + 255) / 256, 256, 0, stream>>>(ids, tok, pos, xf, ROWS * Dd);

  dim3 gD(Dd / BN, ROWS / BM);          // (12,8)  D-output GEMMs
  dim3 gF(FFf / BN, ROWS / BM);         // (48,8)  FF1
  dim3 gS(Tt / BN, Tt / BM, Bb * Hh);   // (8,2,48) scores
  dim3 gC(1, Tt / BM, Bb * Hh);         // (1,2,48) ctx

  for (int i = 0; i < Ll; ++i) {
    layernorm_kernel<<<ROWS, 256, 0, stream>>>(xf, ln1g + i * Dd, ln1b + i * Dd, xln);

    gemm_wmma<<<gD, 256, 0, stream>>>(xln, Dd, Wq + (size_t)i * Dd * Dd, Dd,
                                      nullptr, nullptr, qb, Dd, Dd, Dd, 0);
    gemm_wmma<<<gD, 256, 0, stream>>>(xln, Dd, Wk + (size_t)i * Dd * Dd, Dd,
                                      nullptr, nullptr, kb, Dd, Dd, Dd, 0);
    gemm_wmma<<<gD, 256, 0, stream>>>(xln, Dd, Wv + (size_t)i * Dd * Dd, Dd,
                                      nullptr, nullptr, vb, Dd, Dd, Dd, 0);

    l2norm_qk<<<(ROWS * Hh + 255) / 256, 256, 0, stream>>>(qb, kb, ROWS * Hh);

    attn_scores<<<gS, 256, 0, stream>>>(qb, kb, sc);
    attn_norm_kernel<<<Bb * Hh * Tt, 256, 0, stream>>>(sc);
    attn_ctx<<<gC, 256, 0, stream>>>(sc, vb, cx);

    // x = ctx @ Wo + bo + x   (in-place residual)
    gemm_wmma<<<gD, 256, 0, stream>>>(cx, Dd, Wo + (size_t)i * Dd * Dd, Dd,
                                      bo + (size_t)i * Dd, xf, xf, Dd, Dd, Dd, 2);

    layernorm_kernel<<<ROWS, 256, 0, stream>>>(xf, ln2g + i * Dd, ln2b + i * Dd, xln);

    // mid = gelu(xln @ W1 + b1)
    gemm_wmma<<<gF, 256, 0, stream>>>(xln, Dd, W1 + (size_t)i * Dd * FFf, FFf,
                                      b1 + (size_t)i * FFf, nullptr, mid, FFf,
                                      FFf, Dd, 1);
    // x = mid @ W2 + b2 + x
    gemm_wmma<<<gD, 256, 0, stream>>>(mid, FFf, W2 + (size_t)i * FFf * Dd, Dd,
                                      b2 + (size_t)i * Dd, xf, xf, Dd, Dd, FFf, 2);
  }

  layernorm_kernel<<<ROWS, 256, 0, stream>>>(xf, lnfg, lnfb, xln);

  dim3 gH((Vv + BN - 1) / BN, ROWS / BM);   // (477,8)
  gemm_wmma<<<gH, 256, 0, stream>>>(xln, Dd, headw, Vv, headb, nullptr,
                                    out, Vv, Vv, Dd, 0);
}
